// MO_GNN_large_15109694947580
// MI455X (gfx1250) — matrix operand order repaired and checked by reading, not compiled, not run on hardware
//
#include <hip/hip_runtime.h>
#include <math.h>

// Problem constants (match reference)
#define NN   50000
#define EE   800000
#define INF_ 500
#define HH   256
#define OUTD 40
#define LL   3
#define KP   512   // padded K for IN=500

typedef __attribute__((ext_vector_type(16))) __bf16 v16bf;
typedef __attribute__((ext_vector_type(8)))  __bf16 bf16x8;
typedef __attribute__((ext_vector_type(8)))  float  v8f;

// ---------------------------------------------------------------------------
// softmax of the 5 attention logits
__global__ void k_mask(const float* __restrict__ att, float* __restrict__ mask) {
    float m = att[0];
    for (int i = 1; i < LL + 2; ++i) m = fmaxf(m, att[i]);
    float e[LL + 2], s = 0.f;
    for (int i = 0; i < LL + 2; ++i) { e[i] = expf(att[i] - m); s += e[i]; }
    for (int i = 0; i < LL + 2; ++i) mask[i] = e[i] / s;
}

// fill with 1.0f (self-loop contribution to degree)
__global__ void k_fill1(float* __restrict__ p, long n) {
    long t = (long)blockIdx.x * blockDim.x + threadIdx.x;
    if (t < n) p[t] = 1.0f;
}

// degree scatter for all 3 edge lists: deg[l][dst] += 1
__global__ void k_deg(const int* __restrict__ ei, float* __restrict__ deg) {
    long t = (long)blockIdx.x * blockDim.x + threadIdx.x;
    if (t >= (long)LL * EE) return;
    int l = (int)(t / EE);
    int e = (int)(t % EE);
    int d = ei[(size_t)l * 2 * EE + EE + e];
    atomicAdd(&deg[(size_t)l * NN + d], 1.0f);
}

__global__ void k_rsqrt(float* __restrict__ p, long n) {
    long t = (long)blockIdx.x * blockDim.x + threadIdx.x;
    if (t < n) p[t] = rsqrtf(p[t]);
}

// f32 [rows,K] -> bf16 [rows,Kpad] (zero pad)
__global__ void k_cvtpad(const float* __restrict__ X, __bf16* __restrict__ O,
                         int K, int Kpad, long rows) {
    long t = (long)blockIdx.x * blockDim.x + threadIdx.x;
    if (t >= rows * Kpad) return;
    long i = t / Kpad;
    int  k = (int)(t % Kpad);
    O[t] = (k < K) ? (__bf16)X[i * K + k] : (__bf16)0.0f;
}

// weight transpose-convert: W f32 [K,Nw] row-major -> WT bf16 [Nt,Kpad]
__global__ void k_wt(const float* __restrict__ W, __bf16* __restrict__ WT,
                     int K, int Kpad, int Nw, int Nt) {
    long t = (long)blockIdx.x * blockDim.x + threadIdx.x;
    if (t >= (long)Nt * Kpad) return;
    int n = (int)(t / Kpad);
    int k = (int)(t % Kpad);
    float v = (k < K && n < Nw) ? W[(size_t)k * Nw + n] : 0.0f;
    WT[t] = (__bf16)v;
}

// plain f32 -> bf16
__global__ void k_f2bf(const float* __restrict__ A, __bf16* __restrict__ O, long n) {
    long t = (long)blockIdx.x * blockDim.x + threadIdx.x;
    if (t < n) O[t] = (__bf16)A[t];
}

// ---------------------------------------------------------------------------
// WMMA GEMM: C[M,ncols] = A[M,K](bf16) * B^T  with B stored as WT[n][k] bf16.
// One wave -> 16x64 output tile (4 v8f accumulators).
// A fragment (contiguous k order chosen here, same as B): lane m (0-15) holds K 0..15,
// lane m+16 holds K 16..31 of row m.  B fragment: lane n (0-15) holds
// K 0..15 of column n; lane n+16 holds K 16..31 of column n (contiguous in WT).
__global__ void __launch_bounds__(256) k_gemm_bf16(
    const __bf16* __restrict__ A, const __bf16* __restrict__ B,
    float* __restrict__ C,
    int mtiles, int ntiles, int K, int lda, int ldb, int ldc, int ncols)
{
    int wave = blockIdx.x * (blockDim.x >> 5) + (threadIdx.x >> 5);
    if (wave >= mtiles * ntiles) return;
    const int lane  = threadIdx.x & 31;
    const int mt    = wave / ntiles;
    const int nt    = wave % ntiles;

    const int arow  = mt * 16 + (lane & 15);
    const int ahalf = (lane >> 4) * 16;          // element offset within K-step
    const __bf16* ap = A + (size_t)arow * lda + ahalf;

    const int bcol  = nt * 64 + (lane & 15);
    const int bkoff = (lane >> 4) * 16;
    const __bf16* bp0 = B + (size_t)(bcol +  0) * ldb + bkoff;
    const __bf16* bp1 = B + (size_t)(bcol + 16) * ldb + bkoff;
    const __bf16* bp2 = B + (size_t)(bcol + 32) * ldb + bkoff;
    const __bf16* bp3 = B + (size_t)(bcol + 48) * ldb + bkoff;

    v8f c0 = {}, c1 = {}, c2 = {}, c3 = {};
    for (int k0 = 0; k0 < K; k0 += 32) {
        bf16x8 alo = *(const bf16x8*)(ap + k0);
        bf16x8 ahi = *(const bf16x8*)(ap + k0 + 8);
        v16bf a;
#pragma unroll
        for (int i = 0; i < 8; ++i) { a[i] = alo[i]; a[i + 8] = ahi[i]; }
        v16bf b0 = *(const v16bf*)(bp0 + k0);
        v16bf b1 = *(const v16bf*)(bp1 + k0);
        v16bf b2 = *(const v16bf*)(bp2 + k0);
        v16bf b3 = *(const v16bf*)(bp3 + k0);
        c0 = __builtin_amdgcn_wmma_f32_16x16x32_bf16(false, a, false, b0, (short)0, c0, false, false);
        c1 = __builtin_amdgcn_wmma_f32_16x16x32_bf16(false, a, false, b1, (short)0, c1, false, false);
        c2 = __builtin_amdgcn_wmma_f32_16x16x32_bf16(false, a, false, b2, (short)0, c2, false, false);
        c3 = __builtin_amdgcn_wmma_f32_16x16x32_bf16(false, a, false, b3, (short)0, c3, false, false);
    }

    // C layout: lane L element v -> row = mt*16 + v + (L>=16 ? 8 : 0), col = L&15
    const int crow = mt * 16 + ((lane >> 4) ? 8 : 0);
    const int ccol = lane & 15;
    const int cb   = nt * 64 + ccol;
#pragma unroll
    for (int v = 0; v < 8; ++v) {
        size_t rb = (size_t)(crow + v) * ldc;
        if (cb      < ncols) C[rb + cb     ] = c0[v];
        if (cb + 16 < ncols) C[rb + cb + 16] = c1[v];
        if (cb + 32 < ncols) C[rb + cb + 32] = c2[v];
        if (cb + 48 < ncols) C[rb + cb + 48] = c3[v];
    }
}

// ---------------------------------------------------------------------------
// S[i][c] = HW[i][c] * dinv[i]^2  (self-loop edge of symmetric-norm GCN)
__global__ void k_selfloop(const float* __restrict__ HW, float* __restrict__ S,
                           const float* __restrict__ dinv, long n) {
    long t = (long)blockIdx.x * blockDim.x + threadIdx.x;
    if (t >= n) return;
    long i = t >> 8;
    float di = dinv[i];
    S[t] = HW[t] * di * di;
}

// edge scatter: 64 threads per edge, float4 per thread, f32 atomics (L2-resident)
__global__ void k_scatter(const float* __restrict__ HW, float* __restrict__ S,
                          const int* __restrict__ src, const int* __restrict__ dst,
                          const float* __restrict__ dinv, long E) {
    long t = (long)blockIdx.x * blockDim.x + threadIdx.x;
    long e = t >> 6;
    if (e >= E) return;
    int c = (int)(t & 63) * 4;
    int s = src[e], d = dst[e];
    float nrm = dinv[s] * dinv[d];
    const float4 v = *(const float4*)(HW + (size_t)s * HH + c);
    float* o = S + (size_t)d * HH + c;
    atomicAdd(o + 0, v.x * nrm);
    atomicAdd(o + 1, v.y * nrm);
    atomicAdd(o + 2, v.z * nrm);
    atomicAdd(o + 3, v.w * nrm);
}

// relu(S + b) -> bf16 (feeds next GEMM)
__global__ void k_relu_bias_bf16(const float* __restrict__ S, const float* __restrict__ b,
                                 __bf16* __restrict__ O, long n) {
    long t = (long)blockIdx.x * blockDim.x + threadIdx.x;
    if (t >= n) return;
    int c = (int)(t & (HH - 1));
    O[t] = (__bf16)fmaxf(S[t] + b[c], 0.0f);
}

// ACC (set or add) = relu(S + b) * mask[mi]
__global__ void k_relu_bias_scale(const float* __restrict__ S, const float* __restrict__ b,
                                  const float* __restrict__ maskp, int mi,
                                  float* __restrict__ ACC, int add, long n) {
    long t = (long)blockIdx.x * blockDim.x + threadIdx.x;
    if (t >= n) return;
    int c = (int)(t & (HH - 1));
    float v = fmaxf(S[t] + b[c], 0.0f) * maskp[mi];
    ACC[t] = add ? (ACC[t] + v) : v;
}

// out = log_softmax(out + fc1_b) over the 40 classes, per row
__global__ void k_finish(float* __restrict__ out, const float* __restrict__ bias, int n) {
    int i = blockIdx.x * blockDim.x + threadIdx.x;
    if (i >= n) return;
    float v[OUTD];
    float m = -3.4e38f;
    float* row = out + (size_t)i * OUTD;
#pragma unroll
    for (int c = 0; c < OUTD; ++c) { v[c] = row[c] + bias[c]; m = fmaxf(m, v[c]); }
    float s = 0.f;
#pragma unroll
    for (int c = 0; c < OUTD; ++c) s += expf(v[c] - m);
    float lg = logf(s);
#pragma unroll
    for (int c = 0; c < OUTD; ++c) row[c] = v[c] - m - lg;
}

// ---------------------------------------------------------------------------
extern "C" void kernel_launch(void* const* d_in, const int* in_sizes, int n_in,
                              void* d_out, int out_size, void* d_ws, size_t ws_size,
                              hipStream_t stream) {
    const float* x       = (const float*)d_in[0];
    const int*   ei      = (const int*)  d_in[1];
    const float* att     = (const float*)d_in[2];
    const float* W_mlp   = (const float*)d_in[3];
    const float* b_mlp   = (const float*)d_in[4];
    const float* W_mlp2  = (const float*)d_in[5];
    const float* b_mlp2  = (const float*)d_in[6];
    const float* W_init  = (const float*)d_in[7];
    const float* b_init  = (const float*)d_in[8];
    const float* W_init2 = (const float*)d_in[9];
    const float* b_init2 = (const float*)d_in[10];
    const float* W_convs = (const float*)d_in[11];
    const float* b_convs = (const float*)d_in[12];
    const float* fc1_W   = (const float*)d_in[13];
    const float* fc1_b   = (const float*)d_in[14];
    float* out = (float*)d_out;

    char* ws = (char*)d_ws;
    size_t off = 0;
    auto carve = [&](size_t bytes) { size_t o = off; off += (bytes + 255) & ~(size_t)255; return o; };

    __bf16* XB   = (__bf16*)(ws + carve((size_t)NN * KP * 2));   // x bf16, padded K
    __bf16* HB   = (__bf16*)(ws + carve((size_t)NN * HH * 2));   // hidden bf16
    float*  HW   = (float*) (ws + carve((size_t)NN * HH * 4));   // GEMM output
    float*  S    = (float*) (ws + carve((size_t)NN * HH * 4));   // scatter accumulator
    float*  ACC  = (float*) (ws + carve((size_t)NN * HH * 4));   // running sum
    __bf16* WT   = (__bf16*)(ws + carve((size_t)KP * HH * 2));   // weight^T bf16
    __bf16* WTF  = (__bf16*)(ws + carve((size_t)64 * HH * 2));   // fc1 weight^T (padded to 64)
    float*  DINV = (float*) (ws + carve((size_t)LL * NN * 4));   // per-layer rsqrt(deg)
    float*  MASK = (float*) (ws + carve(8 * 4));

    auto cdiv = [](long a, long b) { return (int)((a + b - 1) / b); };
    const long NHt = (long)NN * HH;

    auto gemm = [&](const __bf16* A, const __bf16* B, float* C,
                    int M, int Npad, int K, int lda, int ldb, int ldc, int ncols) {
        int mt = M / 16, nt = Npad / 64;
        k_gemm_bf16<<<cdiv((long)mt * nt, 8), 256, 0, stream>>>(A, B, C, mt, nt, K, lda, ldb, ldc, ncols);
    };
    auto gcn_agg = [&](int layer) {  // HW -> S with layer's edges + self loops
        const float* dv = DINV + (size_t)layer * NN;
        k_selfloop<<<cdiv(NHt, 256), 256, 0, stream>>>(HW, S, dv, NHt);
        const int* sp = ei + (size_t)layer * 2 * EE;
        const int* dp = sp + EE;
        k_scatter<<<cdiv((long)EE * 64, 256), 256, 0, stream>>>(HW, S, sp, dp, dv, EE);
    };

    // 0) attention mask + degrees
    k_mask<<<1, 1, 0, stream>>>(att, MASK);
    k_fill1<<<cdiv((long)LL * NN, 256), 256, 0, stream>>>(DINV, (long)LL * NN);
    k_deg<<<cdiv((long)LL * EE, 256), 256, 0, stream>>>(ei, DINV);
    k_rsqrt<<<cdiv((long)LL * NN, 256), 256, 0, stream>>>(DINV, (long)LL * NN);

    // 1) x -> bf16 padded
    k_cvtpad<<<cdiv((long)NN * KP, 256), 256, 0, stream>>>(x, XB, INF_, KP, NN);

    // 2) init branch (two-hop GCN on last edge list, mask[4])
    k_wt<<<cdiv((long)HH * KP, 256), 256, 0, stream>>>(W_init, WT, INF_, KP, HH, HH);
    gemm(XB, WT, HW, NN, HH, KP, KP, KP, HH, HH);
    gcn_agg(LL - 1);
    k_relu_bias_bf16<<<cdiv(NHt, 256), 256, 0, stream>>>(S, b_init, HB, NHt);

    k_wt<<<cdiv((long)HH * HH, 256), 256, 0, stream>>>(W_init2, WT, HH, HH, HH, HH);
    gemm(HB, WT, HW, NN, HH, HH, HH, HH, HH, HH);
    gcn_agg(LL - 1);
    k_relu_bias_scale<<<cdiv(NHt, 256), 256, 0, stream>>>(S, b_init2, MASK, 4, ACC, 0, NHt);

    // 3) conv layers i = 0..2, mask[i]
    for (int i = 0; i < LL; ++i) {
        k_wt<<<cdiv((long)HH * KP, 256), 256, 0, stream>>>(W_convs + (size_t)i * INF_ * HH, WT, INF_, KP, HH, HH);
        gemm(XB, WT, HW, NN, HH, KP, KP, KP, HH, HH);
        gcn_agg(i);
        k_relu_bias_scale<<<cdiv(NHt, 256), 256, 0, stream>>>(S, b_convs + (size_t)i * HH, MASK, i, ACC, 1, NHt);
    }

    // 4) MLP branch, mask[3]
    k_wt<<<cdiv((long)HH * KP, 256), 256, 0, stream>>>(W_mlp, WT, INF_, KP, HH, HH);
    gemm(XB, WT, HW, NN, HH, KP, KP, KP, HH, HH);
    k_relu_bias_bf16<<<cdiv(NHt, 256), 256, 0, stream>>>(HW, b_mlp, HB, NHt);

    k_wt<<<cdiv((long)HH * HH, 256), 256, 0, stream>>>(W_mlp2, WT, HH, HH, HH, HH);
    gemm(HB, WT, HW, NN, HH, HH, HH, HH, HH, HH);
    k_relu_bias_scale<<<cdiv(NHt, 256), 256, 0, stream>>>(HW, b_mlp2, MASK, 3, ACC, 1, NHt);

    // 5) fc1 + log_softmax
    k_f2bf<<<cdiv(NHt, 256), 256, 0, stream>>>(ACC, HB, NHt);
    k_wt<<<cdiv((long)64 * HH, 256), 256, 0, stream>>>(fc1_W, WTF, HH, HH, OUTD, 64);
    gemm(HB, WTF, out, NN, 64, HH, HH, HH, OUTD, OUTD);
    k_finish<<<cdiv(NN, 256), 256, 0, stream>>>(out, fc1_b, NN);
}
